// PriorNetwork_88201448391041
// MI455X (gfx1250) — matrix operand order repaired and checked by reading, not compiled, not run on hardware
//
#include <hip/hip_runtime.h>
#include <hip/hip_bf16.h>
#include <math.h>

typedef __attribute__((ext_vector_type(16))) _Float16 v16h;
typedef __attribute__((ext_vector_type(8)))  _Float16 v8h;
typedef __attribute__((ext_vector_type(8)))  float    v8f;

#define DIM       768
#define HEADS     12
#define DIM_HEAD  64
#define DEPTH     6
#define NTOK      257
#define SEQ       515          // 257 brain + 1 time + 257 image
#define BATCH     8
#define ROWS      (BATCH*SEQ)  // 4120
#define KVLEN     516          // null + 515
#define KVPAD     544          // padded to multiple of 32
#define FFI       3072
#define NQT       33           // ceil(515/16)

__device__ __forceinline__ v16h cat16(v8h a, v8h b) {
  return __builtin_shufflevector(a, b, 0,1,2,3,4,5,6,7,8,9,10,11,12,13,14,15);
}
__device__ __forceinline__ v8f wmma32(v16h a, v16h b, v8f c) {
  return __builtin_amdgcn_wmma_f32_16x16x32_f16(false, a, false, b, (short)0, c, false, false);
}
// LDS byte offset of a __shared__ object (generic-space low 32 bits == LDS offset)
__device__ __forceinline__ unsigned lds_addr_of(const void* p) {
  return (unsigned)(reinterpret_cast<uintptr_t>(p));
}

// ---------------------------------------------------------------------------
// WMMA GEMM: C[M,N] = A[M,K](f16,row-major) * B[K,N](f32,row-major)
// Block tile 128x64, BK=32, 8 waves (4x2), each wave 32x32 (4 WMMAs/step).
// A tile is DMA'd global->LDS with GLOBAL_LOAD_ASYNC_TO_LDS_B128 (ASYNCcnt),
// overlapped with the VALU f32->f16 transpose staging of B; fenced with
// s_wait_asynccnt before the barrier. Fragments load as ds_load_b128 in the
// ISA-mandated WMMA VGPR layouts.
// ---------------------------------------------------------------------------
__global__ __launch_bounds__(256) void wmma_gemm(
    const _Float16* __restrict__ A, const float* __restrict__ B,
    void* __restrict__ Cout, int M, int N, int K, int out_f16)
{
  __shared__ _Float16 As[128][40];  // 32 + pad(8): conflict-free fragment reads
  __shared__ _Float16 Bt[64][40];   // transposed: Bt[n][k]
  const int tid  = threadIdx.x;
  const int lane = tid & 31;
  const int w    = tid >> 5;
  const int wrow = w >> 1, wcol = w & 1;     // 4 x 2 waves
  const int m0 = blockIdx.y * 128;
  const int n0 = blockIdx.x * 64;

  v8f acc00 = {0,0,0,0,0,0,0,0};
  v8f acc01 = acc00, acc10 = acc00, acc11 = acc00;

  // per-thread A staging slot: row = tid/2 (0..127), col = 0 or 16 (two b128s)
  const int ar = tid >> 1;
  const int ac = (tid & 1) * 16;
  const unsigned lA0 = lds_addr_of(&As[ar][ac]);
  const unsigned lA1 = lds_addr_of(&As[ar][ac + 8]);

  for (int k0 = 0; k0 < K; k0 += 32) {
    { // --- A tile 128x32 f16 via async DMA to LDS ---
      int gm = m0 + ar;
      if (gm < M) {   // OOB rows keep garbage; their C rows are never written
        unsigned long long g0 =
            (unsigned long long)(uintptr_t)(A + (size_t)gm * K + k0 + ac);
        asm volatile("global_load_async_to_lds_b128 %0, %1, off"
                     :: "v"(lA0), "v"(g0) : "memory");
        unsigned long long g1 = g0 + 16ull;
        asm volatile("global_load_async_to_lds_b128 %0, %1, off"
                     :: "v"(lA1), "v"(g1) : "memory");
      }
    }
    { // --- B tile 32x64: f32 -> f16, stored transposed ---
      int kk = tid >> 3, n8 = (tid & 7) * 8;
      const float* bp = B + (size_t)(k0 + kk) * N + n0 + n8;
      #pragma unroll
      for (int j = 0; j < 8; ++j) Bt[n8 + j][kk] = (_Float16)bp[j];
      if (k0 + 32 < K) __builtin_prefetch(bp + (size_t)32 * N, 0, 1);
    }
    asm volatile("s_wait_asynccnt 0x0" ::: "memory");
    __syncthreads();

    // A fragments: lanes 0-15 rows M with K 0-7/16-23, lanes 16-31 K 8-15/24-31
    int kh8 = (lane >> 4) * 8;
    int ml  = wrow * 32 + (lane & 15);
    v16h af0 = cat16(*(const v8h*)&As[ml][kh8],      *(const v8h*)&As[ml][16 + kh8]);
    v16h af1 = cat16(*(const v8h*)&As[ml + 16][kh8], *(const v8h*)&As[ml + 16][16 + kh8]);

    // B fragments: lane = N column, lanes 0-15 K 0-7/16-23, lanes 16-31 K 8-15/24-31
    int nl = wcol * 32 + (lane & 15);
    int kb = (lane >> 4) * 8;
    v16h bf0 = cat16(*(const v8h*)&Bt[nl][kb],      *(const v8h*)&Bt[nl][kb + 16]);
    v16h bf1 = cat16(*(const v8h*)&Bt[nl + 16][kb], *(const v8h*)&Bt[nl + 16][kb + 16]);

    acc00 = wmma32(af0, bf0, acc00);
    acc01 = wmma32(af0, bf1, acc01);
    acc10 = wmma32(af1, bf0, acc10);
    acc11 = wmma32(af1, bf1, acc11);
    __syncthreads();
  }

  // C layout: VGPR r -> row = r + 8*(lane>=16), col = lane&15
  int hs = lane >> 4, nc = lane & 15;
  #pragma unroll
  for (int r = 0; r < 8; ++r) {
    int row0 = m0 + wrow * 32 + r + hs * 8;
    int row1 = row0 + 16;
    size_t cb = (size_t)n0 + wcol * 32 + nc;
    if (out_f16) {
      _Float16* C = (_Float16*)Cout;
      if (row0 < M) { C[(size_t)row0 * N + cb]      = (_Float16)acc00[r];
                      C[(size_t)row0 * N + cb + 16] = (_Float16)acc01[r]; }
      if (row1 < M) { C[(size_t)row1 * N + cb]      = (_Float16)acc10[r];
                      C[(size_t)row1 * N + cb + 16] = (_Float16)acc11[r]; }
    } else {
      float* C = (float*)Cout;
      if (row0 < M) { C[(size_t)row0 * N + cb]      = acc00[r];
                      C[(size_t)row0 * N + cb + 16] = acc01[r]; }
      if (row1 < M) { C[(size_t)row1 * N + cb]      = acc10[r];
                      C[(size_t)row1 * N + cb + 16] = acc11[r]; }
    }
  }
}

// ---------------------------------------------------------------------------
// Relative position bias (T5-style buckets), emb is [32][HEADS]
// ---------------------------------------------------------------------------
__device__ __forceinline__ float relpos_bias(const float* __restrict__ emb,
                                             int h, int i, int j) {
  int nn = i - j; if (nn < 0) nn = 0;
  int bucket;
  if (nn < 16) bucket = nn;
  else {
    float lf = 16.0f + (__logf((float)nn * (1.0f/16.0f)) / __logf(8.0f)) * 16.0f;
    int lb = (int)lf; bucket = lb < 31 ? lb : 31;
  }
  return emb[bucket * HEADS + h];
}

// ---------------------------------------------------------------------------
// Flash attention, one wave per (b, head, 16-query tile). MQA: K/V shared.
// q: [B][SEQ][HEADS*64] f16 (roped+l2norm*4), k: [B][KVPAD][64] f16 (same),
// vt: [B][64][KVPAD] f16 (transposed V), o: [B][SEQ][HEADS*64] f16
// ---------------------------------------------------------------------------
__global__ __launch_bounds__(256) void attn_kernel(
    const _Float16* __restrict__ q, const _Float16* __restrict__ kf,
    const _Float16* __restrict__ vt, const float* __restrict__ emb,
    _Float16* __restrict__ o)
{
  __shared__ _Float16 Plds[8][16][32];   // per-wave P tile (C-layout -> A-layout)
  const int lane = threadIdx.x & 31;
  const int w    = threadIdx.x >> 5;
  const int gw   = blockIdx.x * 8 + w;   // grid sized exactly B*HEADS*NQT/8
  const int qt = gw % NQT;
  const int h  = (gw / NQT) % HEADS;
  const int b  = gw / (NQT * HEADS);

  const int lh = lane >> 4, ln = lane & 15;
  const int kh8 = lh * 8;

  // Q fragments (K-dim = head dim, two halves of 32)
  int qi_l = qt * 16 + ln; int qi_c = qi_l < SEQ ? qi_l : SEQ - 1;
  const _Float16* qp = q + ((size_t)(b * SEQ + qi_c)) * DIM + h * DIM_HEAD;
  v16h aq0 = cat16(*(const v8h*)(qp + lh * 16),      *(const v8h*)(qp + 8 + lh * 16));
  v16h aq1 = cat16(*(const v8h*)(qp + 32 + lh * 16), *(const v8h*)(qp + 40 + lh * 16));

  float mst[8], lst[8];
  v8f O0 = {0,0,0,0,0,0,0,0}, O1 = O0, O2 = O0, O3 = O0;
  #pragma unroll
  for (int r = 0; r < 8; ++r) { mst[r] = -1e30f; lst[r] = 0.0f; }

  int numkv = qt * 16 + 17; if (numkv > KVLEN) numkv = KVLEN;   // shifted-causal bound
  const _Float16* kbase = kf + (size_t)b * KVPAD * 64;
  const _Float16* vbase = vt + (size_t)b * 64 * KVPAD;

  for (int kv0 = 0; kv0 < numkv; kv0 += 32) {
    // ---- S = Q * K^T over 32 keys (two 16-key subtiles) ----
    v8f S0 = {0,0,0,0,0,0,0,0}, S1 = S0;
    {
      const _Float16* kp = kbase + (size_t)(kv0 + ln) * 64 + lh * 16;
      S0 = wmma32(aq0, cat16(*(const v8h*)kp,        *(const v8h*)(kp + 8)),  S0);
      S0 = wmma32(aq1, cat16(*(const v8h*)(kp + 32), *(const v8h*)(kp + 40)), S0);
    }
    {
      const _Float16* kp = kbase + (size_t)(kv0 + 16 + ln) * 64 + lh * 16;
      S1 = wmma32(aq0, cat16(*(const v8h*)kp,        *(const v8h*)(kp + 8)),  S1);
      S1 = wmma32(aq1, cat16(*(const v8h*)(kp + 32), *(const v8h*)(kp + 40)), S1);
    }
    // ---- bias + mask (padding >= KVLEN and j > i+1 masked) ----
    #pragma unroll
    for (int r = 0; r < 8; ++r) {
      int qi = qt * 16 + r + lh * 8;
      int j0 = kv0 + ln, j1 = j0 + 16;
      float s0 = S0[r] + relpos_bias(emb, h, qi, j0);
      float s1 = S1[r] + relpos_bias(emb, h, qi, j1);
      if (j0 > qi + 1 || j0 >= KVLEN) s0 = -1e30f;
      if (j1 > qi + 1 || j1 >= KVLEN) s1 = -1e30f;
      S0[r] = s0; S1[r] = s1;
    }
    // ---- online softmax (row = 16 lanes of one half, butterfly reduce) ----
    float sc[8];
    #pragma unroll
    for (int r = 0; r < 8; ++r) {
      float mx = fmaxf(S0[r], S1[r]);
      mx = fmaxf(mx, __shfl_xor(mx, 1)); mx = fmaxf(mx, __shfl_xor(mx, 2));
      mx = fmaxf(mx, __shfl_xor(mx, 4)); mx = fmaxf(mx, __shfl_xor(mx, 8));
      float mn = fmaxf(mst[r], mx);
      sc[r] = __expf(mst[r] - mn); mst[r] = mn;
      float p0 = __expf(S0[r] - mn), p1 = __expf(S1[r] - mn);
      float rs = p0 + p1;
      rs += __shfl_xor(rs, 1); rs += __shfl_xor(rs, 2);
      rs += __shfl_xor(rs, 4); rs += __shfl_xor(rs, 8);
      lst[r] = lst[r] * sc[r] + rs;
      Plds[w][r + lh * 8][ln]      = (_Float16)p0;
      Plds[w][r + lh * 8][16 + ln] = (_Float16)p1;
    }
    #pragma unroll
    for (int r = 0; r < 8; ++r) { O0[r]*=sc[r]; O1[r]*=sc[r]; O2[r]*=sc[r]; O3[r]*=sc[r]; }

    asm volatile("s_wait_dscnt 0" ::: "memory");   // P stores visible to our reads

    // ---- O += P(16x32) * V(32x64): P as A-fragment from LDS, V^T contiguous ----
    v16h pf = cat16(*(const v8h*)&Plds[w][ln][lh * 16], *(const v8h*)&Plds[w][ln][lh * 16 + 8]);
    const _Float16* vp = vbase + kv0 + lh * 16;
    {
      const _Float16* v0p = vp + (size_t)(ln)      * KVPAD;
      const _Float16* v1p = vp + (size_t)(16 + ln) * KVPAD;
      const _Float16* v2p = vp + (size_t)(32 + ln) * KVPAD;
      const _Float16* v3p = vp + (size_t)(48 + ln) * KVPAD;
      O0 = wmma32(pf, cat16(*(const v8h*)v0p, *(const v8h*)(v0p + 8)), O0);
      O1 = wmma32(pf, cat16(*(const v8h*)v1p, *(const v8h*)(v1p + 8)), O1);
      O2 = wmma32(pf, cat16(*(const v8h*)v2p, *(const v8h*)(v2p + 8)), O2);
      O3 = wmma32(pf, cat16(*(const v8h*)v3p, *(const v8h*)(v3p + 8)), O3);
    }
  }

  #pragma unroll
  for (int r = 0; r < 8; ++r) {
    int qi = qt * 16 + r + lh * 8;
    if (qi < SEQ) {
      float inv = 1.0f / lst[r];
      size_t base = ((size_t)(b * SEQ + qi)) * DIM + h * DIM_HEAD + ln;
      o[base]      = (_Float16)(O0[r] * inv);
      o[base + 16] = (_Float16)(O1[r] * inv);
      o[base + 32] = (_Float16)(O2[r] * inv);
      o[base + 48] = (_Float16)(O3[r] * inv);
    }
  }
}

// ---------------------------------------------------------------------------
// LayerNorm variants (one 768-wide row per 256-thread block)
// ---------------------------------------------------------------------------
__device__ __forceinline__ float block_reduce_sum(float v, float* red) {
  int tid = threadIdx.x;
  red[tid] = v; __syncthreads();
  for (int off = 128; off > 0; off >>= 1) {
    if (tid < off) red[tid] += red[tid + off];
    __syncthreads();
  }
  float r = red[0]; __syncthreads();
  return r;
}

__global__ __launch_bounds__(256) void ln_f16_kernel(
    const float* __restrict__ x, const float* __restrict__ g, _Float16* __restrict__ out)
{
  __shared__ float red[256];
  int row = blockIdx.x, tid = threadIdx.x;
  const float* xr = x + (size_t)row * DIM;
  float v0 = xr[tid], v1 = xr[tid + 256], v2 = xr[tid + 512];
  float mu = block_reduce_sum(v0 + v1 + v2, red) * (1.0f / DIM);
  float d0 = v0 - mu, d1 = v1 - mu, d2 = v2 - mu;
  float var = block_reduce_sum(d0*d0 + d1*d1 + d2*d2, red) * (1.0f / DIM);
  float is = rsqrtf(var + 1e-5f);
  _Float16* or_ = out + (size_t)row * DIM;
  or_[tid]       = (_Float16)(d0 * is * g[tid]);
  or_[tid + 256] = (_Float16)(d1 * is * g[tid + 256]);
  or_[tid + 512] = (_Float16)(d2 * is * g[tid + 512]);
}

__global__ __launch_bounds__(256) void ln_res_kernel(      // x += ln(y)*g
    const float* __restrict__ y, const float* __restrict__ g, float* __restrict__ x)
{
  __shared__ float red[256];
  int row = blockIdx.x, tid = threadIdx.x;
  const float* yr = y + (size_t)row * DIM;
  float v0 = yr[tid], v1 = yr[tid + 256], v2 = yr[tid + 512];
  float mu = block_reduce_sum(v0 + v1 + v2, red) * (1.0f / DIM);
  float d0 = v0 - mu, d1 = v1 - mu, d2 = v2 - mu;
  float var = block_reduce_sum(d0*d0 + d1*d1 + d2*d2, red) * (1.0f / DIM);
  float is = rsqrtf(var + 1e-5f);
  float* xr = x + (size_t)row * DIM;
  xr[tid]       += d0 * is * g[tid];
  xr[tid + 256] += d1 * is * g[tid + 256];
  xr[tid + 512] += d2 * is * g[tid + 512];
}

__global__ __launch_bounds__(256) void ln_stable_kernel(   // ln(x/max(x))*g -> f16
    const float* __restrict__ x, const float* __restrict__ g, _Float16* __restrict__ out)
{
  __shared__ float red[256];
  int row = blockIdx.x, tid = threadIdx.x;
  const float* xr = x + (size_t)row * DIM;
  float v0 = xr[tid], v1 = xr[tid + 256], v2 = xr[tid + 512];
  red[tid] = fmaxf(fmaxf(v0, v1), v2); __syncthreads();
  for (int off = 128; off > 0; off >>= 1) {
    if (tid < off) red[tid] = fmaxf(red[tid], red[tid + off]);
    __syncthreads();
  }
  float mx = red[0]; __syncthreads();
  v0 /= mx; v1 /= mx; v2 /= mx;
  float mu = block_reduce_sum(v0 + v1 + v2, red) * (1.0f / DIM);
  float d0 = v0 - mu, d1 = v1 - mu, d2 = v2 - mu;
  float var = block_reduce_sum(d0*d0 + d1*d1 + d2*d2, red) * (1.0f / DIM);
  float is = rsqrtf(var + 1e-5f);
  _Float16* or_ = out + (size_t)row * DIM;
  or_[tid]       = (_Float16)(d0 * is * g[tid]);
  or_[tid + 256] = (_Float16)(d1 * is * g[tid + 256]);
  or_[tid + 512] = (_Float16)(d2 * is * g[tid + 512]);
}

// ---------------------------------------------------------------------------
// RoPE + l2norm prep kernels
// ---------------------------------------------------------------------------
__global__ __launch_bounds__(256) void prep_q_kernel(
    const float* __restrict__ qg, _Float16* __restrict__ qo)
{
  int idx = blockIdx.x * 256 + threadIdx.x;
  if (idx >= ROWS * HEADS) return;
  int row = idx / HEADS, h = idx % HEADS;
  float fp = (float)(row % SEQ);
  const float* src = qg + (size_t)row * DIM + h * DIM_HEAD;
  float v[DIM_HEAD];
  #pragma unroll
  for (int p = 0; p < 16; ++p) {
    float inv = __expf((float)(2 * p) * (-logf(10000.0f) / 32.0f));
    float ang = fp * inv, c = __cosf(ang), s = __sinf(ang);
    float t1 = src[2 * p], t2 = src[2 * p + 1];
    v[2 * p] = t1 * c - t2 * s;
    v[2 * p + 1] = t1 * s + t2 * c;
  }
  #pragma unroll
  for (int d = 32; d < 64; ++d) v[d] = src[d];
  float ss = 0.0f;
  #pragma unroll
  for (int d = 0; d < 64; ++d) ss += v[d] * v[d];
  float inv = 4.0f / fmaxf(sqrtf(ss), 1e-12f);   // l2norm * sqrt(16)
  _Float16* dst = qo + (size_t)row * DIM + h * DIM_HEAD;
  #pragma unroll
  for (int d = 0; d < 64; ++d) dst[d] = (_Float16)(v[d] * inv);
}

__global__ __launch_bounds__(256) void prep_kv_kernel(
    const float* __restrict__ kvg, const float* __restrict__ nullkv,
    _Float16* __restrict__ kf, _Float16* __restrict__ vt)
{
  int idx = blockIdx.x * 256 + threadIdx.x;
  if (idx >= BATCH * KVPAD) return;
  int b = idx / KVPAD, kv = idx % KVPAD;
  float kvv[64], vv[64];
  if (kv == 0) {                       // null K/V (K gets l2norm below, no rope)
    #pragma unroll
    for (int d = 0; d < 64; ++d) { kvv[d] = nullkv[d]; vv[d] = nullkv[64 + d]; }
  } else if (kv <= SEQ) {
    int j = kv - 1;
    const float* src = kvg + (size_t)(b * SEQ + j) * 128;
    float fp = (float)j;
    #pragma unroll
    for (int p = 0; p < 16; ++p) {
      float inv = __expf((float)(2 * p) * (-logf(10000.0f) / 32.0f));
      float ang = fp * inv, c = __cosf(ang), s = __sinf(ang);
      float t1 = src[2 * p], t2 = src[2 * p + 1];
      kvv[2 * p] = t1 * c - t2 * s;
      kvv[2 * p + 1] = t1 * s + t2 * c;
    }
    #pragma unroll
    for (int d = 32; d < 64; ++d) kvv[d] = src[d];
    #pragma unroll
    for (int d = 0; d < 64; ++d) vv[d] = src[64 + d];
  } else {                             // padding (always masked)
    #pragma unroll
    for (int d = 0; d < 64; ++d) { kvv[d] = 0.0f; vv[d] = 0.0f; }
  }
  float ss = 0.0f;
  #pragma unroll
  for (int d = 0; d < 64; ++d) ss += kvv[d] * kvv[d];
  float inv = 4.0f / fmaxf(sqrtf(ss), 1e-12f);
  #pragma unroll
  for (int d = 0; d < 64; ++d) {
    kf[((size_t)b * KVPAD + kv) * 64 + d] = (_Float16)(kvv[d] * inv);
    vt[((size_t)b * 64 + d) * KVPAD + kv] = (_Float16)vv[d];
  }
}

// ---------------------------------------------------------------------------
// Small elementwise / time-MLP kernels
// ---------------------------------------------------------------------------
__global__ __launch_bounds__(256) void time_pe_kernel(const int* __restrict__ t,
                                                      float* __restrict__ pe)
{
  int idx = blockIdx.x * 256 + threadIdx.x;
  if (idx >= BATCH * DIM) return;
  int i = idx / DIM, c = idx % DIM;
  float tv = (float)t[i];
  const int half = DIM / 2;
  int cc = (c < half) ? c : c - half;
  float f = __expf((float)cc * (-logf(10000.0f) / (float)(half - 1)));
  pe[idx] = (c < half) ? __sinf(tv * f) : __cosf(tv * f);
}

__global__ __launch_bounds__(256) void dense_kernel(
    const float* __restrict__ in, const float* __restrict__ W,
    const float* __restrict__ bias, float* __restrict__ out,
    int B_, int K, int N, int do_silu)
{
  int idx = blockIdx.x * 256 + threadIdx.x;
  if (idx >= B_ * N) return;
  int i = idx / N, o = idx % N;
  const float* ip = in + (size_t)i * K;
  float s = bias[o];
  for (int k = 0; k < K; ++k) s += ip[k] * W[(size_t)k * N + o];
  if (do_silu) s = s / (1.0f + __expf(-s));
  out[idx] = s;
}

__global__ __launch_bounds__(256) void assemble_kernel(
    const float* __restrict__ brain, const float* __restrict__ te,
    const float* __restrict__ image, const float* __restrict__ lq,
    float* __restrict__ x)
{
  int idx = blockIdx.x * 256 + threadIdx.x;
  if (idx >= ROWS * DIM) return;
  int row = idx / DIM, d = idx % DIM;
  int b = row / SEQ, nn = row % SEQ;
  float v;
  if (nn < NTOK)       v = brain[((size_t)b * NTOK + nn) * DIM + d];
  else if (nn == NTOK) v = te[(size_t)b * DIM + d];
  else { int tt = nn - NTOK - 1;
         v = image[((size_t)b * NTOK + tt) * DIM + d] + lq[(size_t)tt * DIM + d]; }
  x[idx] = v;
}

__global__ __launch_bounds__(256) void geglu_kernel(const _Float16* __restrict__ hf,
                                                    _Float16* __restrict__ act)
{
  int idx = blockIdx.x * 256 + threadIdx.x;
  if (idx >= ROWS * FFI) return;
  int row = idx / FFI, c = idx % FFI;
  float a = (float)hf[(size_t)row * (2 * FFI) + c];
  float g = (float)hf[(size_t)row * (2 * FFI) + FFI + c];
  act[idx] = (_Float16)(a * (g / (1.0f + __expf(-g))));
}

__global__ __launch_bounds__(256) void resadd_kernel(float* __restrict__ x,
                                                     const float* __restrict__ s)
{
  int idx = blockIdx.x * 256 + threadIdx.x;
  if (idx >= ROWS * DIM) return;
  x[idx] += s[idx];
}

__global__ __launch_bounds__(256) void gather_kernel(const float* __restrict__ proj,
                                                     float* __restrict__ out)
{
  int idx = blockIdx.x * 256 + threadIdx.x;
  if (idx >= BATCH * NTOK * DIM) return;
  int b = idx / (NTOK * DIM);
  int rem = idx % (NTOK * DIM);
  int t = rem / DIM, d = rem % DIM;
  out[idx] = proj[((size_t)(b * SEQ + (SEQ - NTOK) + t)) * DIM + d];
}

// ---------------------------------------------------------------------------
extern "C" void kernel_launch(void* const* d_in, const int* in_sizes, int n_in,
                              void* d_out, int out_size, void* d_ws, size_t ws_size,
                              hipStream_t stream) {
  const float* image  = (const float*)d_in[0];
  const float* brain  = (const float*)d_in[1];
  const int*   tsteps = (const int*)  d_in[2];
  const float* lq     = (const float*)d_in[3];
  const float* t_w1   = (const float*)d_in[4];
  const float* t_b1   = (const float*)d_in[5];
  const float* t_w2   = (const float*)d_in[6];
  const float* t_b2   = (const float*)d_in[7];
  const float* t_w3   = (const float*)d_in[8];
  const float* t_b3   = (const float*)d_in[9];
  const float* relpos = (const float*)d_in[10];
  const float* attn_norm_g = (const float*)d_in[11];
  const float* Wq     = (const float*)d_in[12];
  const float* Wkv    = (const float*)d_in[13];
  const float* nullkv = (const float*)d_in[14];
  const float* Wo     = (const float*)d_in[15];
  const float* attn_out_g = (const float*)d_in[16];
  const float* ff_norm_g  = (const float*)d_in[17];
  const float* ff_w_in    = (const float*)d_in[18];
  const float* ff_w_out   = (const float*)d_in[19];
  const float* final_g    = (const float*)d_in[20];
  const float* Wproj      = (const float*)d_in[21];

  char* ws = (char*)d_ws;
  size_t off = 0;
  auto take = [&](size_t bytes) { size_t r = off; off += (bytes + 255) & ~(size_t)255; return r; };
  float*    x    = (float*)   (ws + take((size_t)ROWS * DIM * 4));
  _Float16* xn   = (_Float16*)(ws + take((size_t)ROWS * DIM * 2));
  float*    scr  = (float*)   (ws + take((size_t)ROWS * DIM * 4));
  float*    kv32 = (float*)   (ws + take((size_t)ROWS * 128 * 4));
  _Float16* qf   = (_Float16*)(ws + take((size_t)ROWS * DIM * 2));
  _Float16* kf   = (_Float16*)(ws + take((size_t)BATCH * KVPAD * 64 * 2));
  _Float16* vtb  = (_Float16*)(ws + take((size_t)BATCH * 64 * KVPAD * 2));
  _Float16* of   = (_Float16*)(ws + take((size_t)ROWS * DIM * 2));
  _Float16* hf   = (_Float16*)(ws + take((size_t)ROWS * 2 * FFI * 2));
  _Float16* act  = (_Float16*)(ws + take((size_t)ROWS * FFI * 2));
  float*    pe   = (float*)   (ws + take((size_t)BATCH * DIM * 4));
  float*    h1   = (float*)   (ws + take((size_t)BATCH * 1536 * 4));
  float*    h2   = (float*)   (ws + take((size_t)BATCH * 1536 * 4));
  float*    te   = (float*)   (ws + take((size_t)BATCH * DIM * 4));
  (void)ws_size; (void)n_in; (void)in_sizes; (void)out_size;

  const int MYG = (ROWS + 127) / 128;   // 33 M-tiles (128-row GEMM blocks)

  // --- time embedding MLP (tiny) ---
  time_pe_kernel<<<(BATCH * DIM + 255) / 256, 256, 0, stream>>>(tsteps, pe);
  dense_kernel<<<(BATCH * 1536 + 255) / 256, 256, 0, stream>>>(pe, t_w1, t_b1, h1, BATCH, DIM, 1536, 1);
  dense_kernel<<<(BATCH * 1536 + 255) / 256, 256, 0, stream>>>(h1, t_w2, t_b2, h2, BATCH, 1536, 1536, 1);
  dense_kernel<<<(BATCH * DIM + 255) / 256, 256, 0, stream>>>(h2, t_w3, t_b3, te, BATCH, 1536, DIM, 0);

  // --- token assembly ---
  assemble_kernel<<<(ROWS * DIM + 255) / 256, 256, 0, stream>>>(brain, te, image, lq, x);

  for (int l = 0; l < DEPTH; ++l) {
    const float* Wq_l   = Wq   + (size_t)l * DIM * DIM;
    const float* Wkv_l  = Wkv  + (size_t)l * DIM * 128;
    const float* nul_l  = nullkv + (size_t)l * 2 * 64;
    const float* Wo_l   = Wo   + (size_t)l * DIM * DIM;
    const float* ffi_l  = ff_w_in  + (size_t)l * DIM * 2 * FFI;
    const float* ffo_l  = ff_w_out + (size_t)l * FFI * DIM;

    ln_f16_kernel<<<ROWS, 256, 0, stream>>>(x, attn_norm_g + l * DIM, xn);
    wmma_gemm<<<dim3(DIM / 64, MYG), 256, 0, stream>>>(xn, Wq_l, scr, ROWS, DIM, DIM, 0);
    wmma_gemm<<<dim3(128 / 64, MYG), 256, 0, stream>>>(xn, Wkv_l, kv32, ROWS, 128, DIM, 0);
    prep_q_kernel<<<(ROWS * HEADS + 255) / 256, 256, 0, stream>>>(scr, qf);
    prep_kv_kernel<<<(BATCH * KVPAD + 255) / 256, 256, 0, stream>>>(kv32, nul_l, kf, vtb);
    attn_kernel<<<(BATCH * HEADS * NQT) / 8, 256, 0, stream>>>(qf, kf, vtb, relpos, of);
    wmma_gemm<<<dim3(DIM / 64, MYG), 256, 0, stream>>>(of, Wo_l, scr, ROWS, DIM, DIM, 0);
    ln_res_kernel<<<ROWS, 256, 0, stream>>>(scr, attn_out_g + l * DIM, x);

    ln_f16_kernel<<<ROWS, 256, 0, stream>>>(x, ff_norm_g + l * DIM, xn);
    wmma_gemm<<<dim3((2 * FFI) / 64, MYG), 256, 0, stream>>>(xn, ffi_l, hf, ROWS, 2 * FFI, DIM, 1);
    geglu_kernel<<<(ROWS * FFI + 255) / 256, 256, 0, stream>>>(hf, act);
    wmma_gemm<<<dim3(DIM / 64, MYG), 256, 0, stream>>>(act, ffo_l, scr, ROWS, DIM, FFI, 0);
    resadd_kernel<<<(ROWS * DIM + 255) / 256, 256, 0, stream>>>(x, scr);
  }

  ln_stable_kernel<<<ROWS, 256, 0, stream>>>(x, final_g, xn);
  wmma_gemm<<<dim3(DIM / 64, MYG), 256, 0, stream>>>(xn, Wproj, scr, ROWS, DIM, DIM, 0);
  gather_kernel<<<(BATCH * NTOK * DIM + 255) / 256, 256, 0, stream>>>(scr, (float*)d_out);
}
